// ImageAttention_11768210391135
// MI455X (gfx1250) — matrix operand order repaired and checked by reading, not compiled, not run on hardware
//
#include <hip/hip_runtime.h>
#include <stdint.h>

#define B_    16
#define C_    256
#define H_    96
#define W_    256
#define HW_   (H_*W_)
#define KK    11
#define PATCH (KK*KK)   /* 121 */
#define KTOT  256       /* 2*121 padded to 256 */

typedef __attribute__((ext_vector_type(16))) __bf16 v16bf;
typedef __attribute__((ext_vector_type(8)))  float  v8f;

union Frag {
  v16bf v;
  unsigned short u16[16];
  uint4 q[2];
};

__device__ __forceinline__ unsigned short f2bf(float f) {
  union { float f; unsigned int u; } x; x.f = f;
  unsigned int r = x.u + 0x7FFFu + ((x.u >> 16) & 1u);   // round-to-nearest-even
  return (unsigned short)(r >> 16);
}

__device__ __forceinline__ float fast_tanh(float x) {
#if __has_builtin(__builtin_amdgcn_tanhf)
  return __builtin_amdgcn_tanhf(x);       // gfx1250 V_TANH_F32
#else
  float t = __expf(2.0f * x);             // branch-free fallback
  return 1.0f - 2.0f / (t + 1.0f);
#endif
}

// ---- prep: combined conv weights [256][256] bf16 (k<121: Wcov, 121..241: Wcum, rest 0)
__global__ void prep_w(const float* __restrict__ Wcov, const float* __restrict__ Wcum,
                       unsigned short* __restrict__ wcomb) {
  int c = threadIdx.x;
  for (int k = 0; k < KTOT; ++k) {
    float v = 0.f;
    if (k < PATCH)        v = Wcov[c*PATCH + k];
    else if (k < 2*PATCH) v = Wcum[c*PATCH + (k - PATCH)];
    wcomb[c*KTOT + k] = f2bf(v);
  }
}

// ---- prep: qwl[b][c] = { (query @ Wq.T)[b,c] + bq[c] + bcov[c] + bcum[c], Wlog[c] }
__global__ void prep_q(const float* __restrict__ query, const float* __restrict__ Wq,
                       const float* __restrict__ bq, const float* __restrict__ bcov,
                       const float* __restrict__ bcum, const float* __restrict__ Wlog,
                       float2* __restrict__ qwl) {
  int b = blockIdx.x, c = threadIdx.x;
  float acc = 0.f;
  for (int j = 0; j < C_; ++j) acc += query[b*C_ + j] * Wq[c*C_ + j];
  qwl[b*C_ + c] = make_float2(acc + bq[c] + bcov[c] + bcum[c], Wlog[c]);
}

__global__ void count_layouts(const float* __restrict__ layouts, float* __restrict__ counts) {
  __shared__ float red[256];
  int b = blockIdx.x, t = threadIdx.x;
  float s = 0.f;
  for (int i = t; i < HW_; i += 256) s += layouts[b*HW_ + i];
  red[t] = s; __syncthreads();
  for (int k = 128; k > 0; k >>= 1) { if (t < k) red[t] += red[t+k]; __syncthreads(); }
  if (t == 0) counts[b] = red[0];
}

__global__ void copy_state(const float* __restrict__ state, float* __restrict__ out) {
  int i = blockIdx.x*256 + threadIdx.x;
  if (i < B_*512) out[i] = state[i];
}

__global__ void new_cum_k(const float* __restrict__ layouts, const float* __restrict__ cum,
                          float* __restrict__ out) {
  int i = blockIdx.x*256 + threadIdx.x;
  if (i < B_*HW_) out[i] = fminf(layouts[i] + cum[i], 1.0f);
}

// ---- main fused kernel: conv-as-GEMM (bf16 WMMA) + key fusion + tanh + Wlog reduce -> logit
__global__ void __launch_bounds__(256)
fused_conv_logit(const float* __restrict__ key, const float* __restrict__ saw,
                 const float* __restrict__ csaw, const unsigned short* __restrict__ wcomb,
                 const float2* __restrict__ qwl, const float* __restrict__ blog,
                 float* __restrict__ logit_out) {
  __shared__ uint4  wlds[8192];        // 128 KB: full bf16 weight matrix [256][256]
  __shared__ float  win[2][KK][144];   // halo windows for both conv inputs (138 cols used)

  int blk = blockIdx.x;
  int b   = blk / (H_*2);
  int rem = blk % (H_*2);
  int h   = rem >> 1;
  int w0  = (rem & 1) * 128;

  // stage full weight matrix into LDS (one 128 KB read per block)
  for (int i = threadIdx.x; i < 8192; i += 256)
    wlds[i] = ((const uint4*)wcomb)[i];

  // stage 11-row halo windows (zero-padded at image edges)
  for (int i = threadIdx.x; i < 2*KK*138; i += 256) {
    int s   = i / (KK*138);
    int r2  = i % (KK*138);
    int r   = r2 / 138;
    int col = r2 % 138;
    int hh  = h + r - 5;
    int ww  = w0 + col - 5;
    float v = 0.f;
    if (hh >= 0 && hh < H_ && ww >= 0 && ww < W_) {
      const float* g = (s == 0) ? saw : csaw;
      v = g[b*HW_ + hh*W_ + ww];
    }
    win[s][r][col] = v;
  }
  __syncthreads();

  int tid  = threadIdx.x;
  int wave = tid >> 5;
  int lane = tid & 31;
  int n    = lane & 15;      // pixel index within wave tile (B/D column)
  int half = lane >> 4;
  int tp   = wave*16 + n;    // pixel offset within the 128-pixel block tile
  int wglob = w0 + tp;

  // Build B fragments (patch matrix, K x N), held in registers, reused for all channel tiles.
  // B layout (32x16, bf16): VGPR j, lanes 0-15: n=lane, k={2j,2j+1}; lanes 16-31: n=lane-16, k+=16.
  Frag Bf[8];
  #pragma unroll
  for (int s = 0; s < 8; ++s) {
    #pragma unroll
    for (int j = 0; j < 8; ++j) {
      #pragma unroll
      for (int e = 0; e < 2; ++e) {
        int k = s*32 + half*16 + 2*j + e;
        float v = 0.f;
        if (k < PATCH) {
          int dy = k / KK, dx = k % KK;
          v = win[0][dy][tp + dx];
        } else if (k < 2*PATCH) {
          int kk = k - PATCH;
          int dy = kk / KK, dx = kk % KK;
          v = win[1][dy][tp + dx];
        }
        Bf[s].u16[2*j + e] = f2bf(v);
      }
    }
  }

  float bl = blog[0];
  float partial = 0.f;
  int m = lane & 15;                                // A row within channel tile
  const float* keyb = key + (size_t)b*C_*HW_ + (size_t)h*W_ + wglob;
  const float2* qwlb = qwl + b*C_;

  for (int ct = 0; ct < 16; ++ct) {
    int crow = ct*16 + m;
    const uint4* rowp = &wlds[crow*32];    // 32 uint4 per 512B row
    v8f acc = {};
    #pragma unroll
    for (int s = 0; s < 8; ++s) {
      // A layout (16x32, bf16): lanes hold M=lane&15; VGPR j: k=2j+half*16 (contiguous, matches B)
      Frag Af;
      Af.q[0] = rowp[s*4 + 2*half];     // k = s*32 + half*16 + {0..7}
      Af.q[1] = rowp[s*4 + 2*half + 1]; // k = s*32 + half*16 + 8 + {0..7}
      acc = __builtin_amdgcn_wmma_f32_16x16x32_bf16(false, Af.v, false, Bf[s].v,
                                                    (short)0, acc, false, false);
    }
    // Epilogue: D VGPR r -> channel ct*16 + r + half*8, column = pixel n
    #pragma unroll
    for (int r = 0; r < 8; ++r) {
      int c = ct*16 + r + half*8;
      float keyv = keyb[(size_t)c*HW_];
      float2 qw = qwlb[c];
      float f = keyv + qw.x + acc[r];
      partial += qw.y * fast_tanh(f);
    }
  }
  // lanes L and L^16 hold complementary channel halves of the same pixel
  float other = __shfl_xor(partial, 16, 32);
  float lv = partial + other + bl;
  if (half == 0) logit_out[b*HW_ + h*W_ + wglob] = lv;
}

__global__ void softmax_k(const float* __restrict__ logit, const float* __restrict__ mask,
                          float* __restrict__ att) {
  __shared__ float red[256];
  int b = blockIdx.x, t = threadIdx.x;
  float mx = -1e30f;
  for (int i = t; i < HW_; i += 256) {
    float mv = logit[b*HW_ + i] - (1.0f - mask[b*HW_ + i]) * 1e8f;
    mx = fmaxf(mx, mv);
  }
  red[t] = mx; __syncthreads();
  for (int k = 128; k > 0; k >>= 1) { if (t < k) red[t] = fmaxf(red[t], red[t+k]); __syncthreads(); }
  mx = red[0]; __syncthreads();
  float sum = 0.f;
  for (int i = t; i < HW_; i += 256) {
    float mv = logit[b*HW_ + i] - (1.0f - mask[b*HW_ + i]) * 1e8f;
    sum += __expf(mv - mx);
  }
  red[t] = sum; __syncthreads();
  for (int k = 128; k > 0; k >>= 1) { if (t < k) red[t] += red[t+k]; __syncthreads(); }
  sum = red[0];
  float inv = 1.0f / sum;
  for (int i = t; i < HW_; i += 256) {
    float mv = logit[b*HW_ + i] - (1.0f - mask[b*HW_ + i]) * 1e8f;
    att[b*HW_ + i] = __expf(mv - mx) * inv;
  }
}

__global__ void value_reduce(const float* __restrict__ value, const float* __restrict__ layouts,
                             const float* __restrict__ counts, float* __restrict__ outputs) {
  __shared__ float red[256];
  int bc = blockIdx.x;        // b*256 + c
  int b = bc >> 8;
  int t = threadIdx.x;
  const float* v = value + (size_t)bc*HW_;
  const float* l = layouts + (size_t)b*HW_;
  float s = 0.f;
  for (int i = t; i < HW_; i += 256) s += v[i]*l[i];
  red[t] = s; __syncthreads();
  for (int k = 128; k > 0; k >>= 1) { if (t < k) red[t] += red[t+k]; __syncthreads(); }
  if (t == 0) {
    float cnt = counts[b];
    outputs[bc] = (cnt > 0.f) ? red[0] / fmaxf(cnt, 1.0f) : 0.f;
  }
}

extern "C" void kernel_launch(void* const* d_in, const int* in_sizes, int n_in,
                              void* d_out, int out_size, void* d_ws, size_t ws_size,
                              hipStream_t stream) {
  const float* key     = (const float*)d_in[0];
  const float* mask    = (const float*)d_in[1];
  const float* query   = (const float*)d_in[2];
  const float* saw     = (const float*)d_in[3];
  const float* csaw    = (const float*)d_in[4];
  const float* value   = (const float*)d_in[5];
  const float* state   = (const float*)d_in[6];
  const float* layouts = (const float*)d_in[7];
  const float* Wq      = (const float*)d_in[8];
  const float* bq      = (const float*)d_in[9];
  const float* Wcov    = (const float*)d_in[10];
  const float* bcov    = (const float*)d_in[11];
  const float* Wcum    = (const float*)d_in[12];
  const float* bcum    = (const float*)d_in[13];
  const float* Wlog    = (const float*)d_in[14];
  const float* blog    = (const float*)d_in[15];

  float* out         = (float*)d_out;
  float* out_state   = out;                       // 16*512
  float* out_outputs = out + 8192;                // 16*256
  float* out_logit   = out + 12288;               // 16*96*256
  float* out_att     = out + 12288 + B_*HW_;
  float* out_cum     = out + 12288 + 2*B_*HW_;

  unsigned short* wcomb = (unsigned short*)d_ws;              // 131072 B
  float2* qwl   = (float2*)((char*)d_ws + 131072);            // 32768 B
  float* counts = (float*)((char*)d_ws + 164096);             // 64 B (16B-aligned region after qwl)

  prep_w<<<1, 256, 0, stream>>>(Wcov, Wcum, wcomb);
  prep_q<<<B_, 256, 0, stream>>>(query, Wq, bq, bcov, bcum, Wlog, qwl);
  count_layouts<<<B_, 256, 0, stream>>>(layouts, counts);
  copy_state<<<32, 256, 0, stream>>>(state, out_state);
  new_cum_k<<<(B_*HW_ + 255)/256, 256, 0, stream>>>(layouts, csaw, out_cum);
  fused_conv_logit<<<B_*H_*2, 256, 0, stream>>>(key, saw, csaw, wcomb, qwl, blog, out_logit);
  softmax_k<<<B_, 256, 0, stream>>>(out_logit, mask, out_att);
  value_reduce<<<B_*C_, 256, 0, stream>>>(value, layouts, counts, out_outputs);
}
